// QLayer_13099650253498
// MI455X (gfx1250) — matrix operand order repaired and checked by reading, not compiled, not run on hardware
//
#include <hip/hip_runtime.h>

typedef __attribute__((ext_vector_type(16))) __bf16 v16bf;
typedef __attribute__((ext_vector_type(8)))  __bf16 v8bf;
typedef __attribute__((ext_vector_type(4)))  __bf16 v4bf;
typedef __attribute__((ext_vector_type(8)))  float  v8f;

#define T_    4
#define N_    16384
#define DIN   512
#define DH    1024
#define MAXO  128
#define BM    128
#define XP    136   // LDS row stride in halves: 128 + 8 pad (4-bank rotate / row)

// workspace layout (bytes)
#define W1_HALVES (T_*64*16*512)            // [t][ntile:64][ktile:16][512]
#define W2_HALVES (T_*8*32*512)             // [t][ntile:8][ktile:32][512]
#define W2_OFF    ((size_t)W1_HALVES*2)
#define B2_OFF    (W2_OFF + (size_t)W2_HALVES*2)

// ---------------- weight repack: fp32 -> bf16 B-fragment layout ----------------
// B frag (16-bit, KxN = 32x16): lane holds N = lane&15; lanes 0-15 hold K=0..7,16..23,
// lanes 16-31 hold K=8..15,24..31; 16 contiguous halves per lane.
__global__ void pack_w1(const float* __restrict__ W1, __bf16* __restrict__ out) {
    int idx  = blockIdx.x * 256 + threadIdx.x;      // 2,097,152 total
    int j    = idx & 15;
    int lane = (idx >> 4) & 31;
    int ktl  = (idx >> 9) & 15;
    int ntl  = (idx >> 13) & 63;
    int t    = idx >> 19;
    int k = ktl * 32 + (j >> 3) * 16 + (lane >> 4) * 8 + (j & 7);
    int n = ntl * 16 + (lane & 15);
    out[idx] = (__bf16)W1[((size_t)t * DIN + k) * DH + n];
}

__global__ void pack_w2(const float* __restrict__ W20, const float* __restrict__ W21,
                        const float* __restrict__ W22, const float* __restrict__ W23,
                        __bf16* __restrict__ out) {
    int idx  = blockIdx.x * 256 + threadIdx.x;      // 524,288 total
    int j    = idx & 15;
    int lane = (idx >> 4) & 31;
    int ktl  = (idx >> 9) & 31;
    int ntl  = (idx >> 14) & 7;
    int t    = idx >> 17;
    int k = ktl * 32 + (j >> 3) * 16 + (lane >> 4) * 8 + (j & 7);
    int n = ntl * 16 + (lane & 15);
    const float* W2[T_] = {W20, W21, W22, W23};
    const int    osz[T_] = {128, 96, 64, 32};
    float v = (n < osz[t]) ? W2[t][(size_t)k * osz[t] + n] : 0.0f;
    out[idx] = (__bf16)v;
}

__global__ void pack_b2(const float* __restrict__ B20, const float* __restrict__ B21,
                        const float* __restrict__ B22, const float* __restrict__ B23,
                        float* __restrict__ out) {
    int idx = blockIdx.x * 256 + threadIdx.x;       // 512 total
    if (idx >= T_ * MAXO) return;
    int t = idx >> 7, c = idx & 127;
    const float* B2[T_] = {B20, B21, B22, B23};
    const int    osz[T_] = {128, 96, 64, 32};
    out[idx] = (c < osz[t]) ? B2[t][c] : 0.0f;
}

// ---------------- fused 2-layer MLP ----------------
__launch_bounds__(256)
__global__ void fused_mlp(const float* __restrict__ x0, const float* __restrict__ x1,
                          const float* __restrict__ x2, const float* __restrict__ x3,
                          const float* __restrict__ b1,
                          const __bf16* __restrict__ w1p, const __bf16* __restrict__ w2p,
                          const float* __restrict__ b2p, float* __restrict__ out) {
    // Single LDS buffer: during layer1 it holds the 128x128 x-chunk (bf16);
    // between layer1 and layer2 each wave reuses ONLY its own 16-row strip as
    // h-scratch (cross-wave safe: wave w reads/writes rows 16w..16w+15 only).
    __shared__ __bf16 xsm[BM][XP];

    const int t    = blockIdx.y;
    const float* xp = (t == 0) ? x0 : (t == 1) ? x1 : (t == 2) ? x2 : x3;
    const int row0 = blockIdx.x * BM;
    const int tid  = threadIdx.x;
    const int w    = tid >> 5;          // wave 0..7 -> row strip
    const int lane = tid & 31;
    const int lmod = lane & 15;
    const int hk   = (lane >> 4) * 8;   // A-frag K sub-offset (halves)

    v8f zero = {};
    v8f oacc[8];
#pragma unroll
    for (int nt = 0; nt < 8; ++nt) oacc[nt] = zero;

#pragma unroll 1
    for (int panel = 0; panel < 8; ++panel) {       // D_H in 128-wide panels
        v8f hacc[8];
#pragma unroll
        for (int nt = 0; nt < 8; ++nt) hacc[nt] = zero;

#pragma unroll 1
        for (int kc = 0; kc < 4; ++kc) {            // D_IN in 128-wide K chunks
            __syncthreads();
            // stage x[:, kc*128 .. +128) -> bf16 LDS (each thread: 16 float4)
#pragma unroll
            for (int it = 0; it < 16; ++it) {
                int lin = (tid + it * 256) * 4;     // 0..16383
                int r = lin >> 7, c = lin & 127;
                const float4 v = *(const float4*)&xp[(size_t)(row0 + r) * DIN + kc * 128 + c];
                v4bf pk = {(__bf16)v.x, (__bf16)v.y, (__bf16)v.z, (__bf16)v.w};
                *(v4bf*)&xsm[r][c] = pk;
            }
            // warm L0/L2 for the next chunk (wraps to chunk 0 for the next panel)
            __builtin_prefetch(&xp[(size_t)(row0 + (tid >> 1)) * DIN + ((kc + 1) & 3) * 128 + (tid & 1) * 64], 0, 0);
            __syncthreads();

#pragma unroll
            for (int kt = 0; kt < 4; ++kt) {        // 4 x K=32 steps per chunk
                const v8bf a_lo = *(const v8bf*)&xsm[w * 16 + lmod][kt * 32 + hk];
                const v8bf a_hi = *(const v8bf*)&xsm[w * 16 + lmod][kt * 32 + 16 + hk];
                const v16bf a = __builtin_shufflevector(a_lo, a_hi,
                    0, 1, 2, 3, 4, 5, 6, 7, 8, 9, 10, 11, 12, 13, 14, 15);
                const __bf16* wb = w1p +
                    ((((size_t)t * 64 + panel * 8) * 16 + (kc * 4 + kt)) * 512) + lane * 16;
#pragma unroll
                for (int nt = 0; nt < 8; ++nt) {
                    const v16bf b = *(const v16bf*)(wb + (size_t)nt * (16 * 512));
                    hacc[nt] = __builtin_amdgcn_wmma_f32_16x16x32_bf16(
                        false, a, false, b, (short)0, hacc[nt], false, false);
                }
            }
        }

        // bias + ReLU; C-layout -> A-layout transpose via this wave's LDS strip
#pragma unroll
        for (int nt = 0; nt < 8; ++nt) {
            const float bias = b1[(size_t)t * DH + panel * 128 + nt * 16 + lmod];
#pragma unroll
            for (int v = 0; v < 8; ++v) {
                float hv = hacc[nt][v] + bias;
                hv = hv > 0.0f ? hv : 0.0f;
                // C frag: lane -> col (N), vgpr v -> row (M = (lane/16)*8 + v)
                xsm[w * 16 + (hk & 8) + v][nt * 16 + lmod] = (__bf16)hv;
            }
        }

        // layer2 partial: oacc += h_panel[16 x 128] @ W2[panel*128.., 0..127]
#pragma unroll
        for (int kt2 = 0; kt2 < 4; ++kt2) {
            const v8bf a_lo = *(const v8bf*)&xsm[w * 16 + lmod][kt2 * 32 + hk];
            const v8bf a_hi = *(const v8bf*)&xsm[w * 16 + lmod][kt2 * 32 + 16 + hk];
            const v16bf a = __builtin_shufflevector(a_lo, a_hi,
                0, 1, 2, 3, 4, 5, 6, 7, 8, 9, 10, 11, 12, 13, 14, 15);
            const __bf16* wb = w2p +
                (((size_t)t * 8 * 32 + (panel * 4 + kt2)) * 512) + lane * 16;
#pragma unroll
            for (int nt = 0; nt < 8; ++nt) {
                const v16bf b = *(const v16bf*)(wb + (size_t)nt * (32 * 512));
                oacc[nt] = __builtin_amdgcn_wmma_f32_16x16x32_bf16(
                    false, a, false, b, (short)0, oacc[nt], false, false);
            }
        }
    }

    // epilogue: out[t][row][col] = oacc + b2_pad
#pragma unroll
    for (int nt = 0; nt < 8; ++nt) {
        const int col = nt * 16 + lmod;
        const float bias = b2p[t * MAXO + col];
#pragma unroll
        for (int v = 0; v < 8; ++v) {
            const int row = row0 + w * 16 + (lane >> 4) * 8 + v;
            out[((size_t)t * N_ + row) * MAXO + col] = oacc[nt][v] + bias;
        }
    }
}

extern "C" void kernel_launch(void* const* d_in, const int* in_sizes, int n_in,
                              void* d_out, int out_size, void* d_ws, size_t ws_size,
                              hipStream_t stream) {
    const float* x0 = (const float*)d_in[0];
    const float* x1 = (const float*)d_in[1];
    const float* x2 = (const float*)d_in[2];
    const float* x3 = (const float*)d_in[3];
    // d_in[4] = node_type (sorted/blocked) -> structurally unused
    const float* W1 = (const float*)d_in[5];
    const float* b1 = (const float*)d_in[6];
    const float* W2_0 = (const float*)d_in[7];  const float* b2_0 = (const float*)d_in[8];
    const float* W2_1 = (const float*)d_in[9];  const float* b2_1 = (const float*)d_in[10];
    const float* W2_2 = (const float*)d_in[11]; const float* b2_2 = (const float*)d_in[12];
    const float* W2_3 = (const float*)d_in[13]; const float* b2_3 = (const float*)d_in[14];

    __bf16* w1p = (__bf16*)((char*)d_ws);
    __bf16* w2p = (__bf16*)((char*)d_ws + W2_OFF);
    float*  b2p = (float*)((char*)d_ws + B2_OFF);

    pack_w1<<<W1_HALVES / 256, 256, 0, stream>>>(W1, w1p);
    pack_w2<<<W2_HALVES / 256, 256, 0, stream>>>(W2_0, W2_1, W2_2, W2_3, w2p);
    pack_b2<<<2, 256, 0, stream>>>(b2_0, b2_1, b2_2, b2_3, b2p);

    dim3 grid(N_ / BM, T_);
    fused_mlp<<<grid, 256, 0, stream>>>(x0, x1, x2, x3, b1, w1p, w2p, b2p, (float*)d_out);
}
